// SwinBlock1D_83382495084932
// MI455X (gfx1250) — matrix operand order repaired and checked by reading, not compiled, not run on hardware
//
#include <hip/hip_runtime.h>
#include <hip/hip_bf16.h>

// ---------------------------------------------------------------------------
// SwinBlock1D for MI455X (gfx1250, wave32, WMMA + async global->LDS).
// B=8 L=4096 E=1024 H=16 W=128 SHIFT=64 HD=64 MLP=4096  -> bf16 WMMA GEMMs.
// ---------------------------------------------------------------------------

#define BATCH   8
#define SEQL    4096
#define EMBED   1024
#define HEADS   16
#define WIN     128
#define SHIFTA  64
#define HD      64
#define MLP_H   4096
#define NWIN    32              // SEQL / WIN
#define NTOK    (BATCH * SEQL)  // 32768 tokens
#define SCALEA  0.125f          // HD^-0.5

typedef __attribute__((ext_vector_type(16))) __bf16        v16bf;
typedef __attribute__((ext_vector_type(8)))  float         v8f;
typedef __attribute__((ext_vector_type(4)))  unsigned int  u32x4;

union FragAB { v16bf v; u32x4 q[2]; };
union FragC  { v8f   v; float f[8]; };

__device__ __forceinline__ unsigned short f2bf(float f) {
    unsigned int u = __builtin_bit_cast(unsigned int, f);
    u += 0x7fffu + ((u >> 16) & 1u);            // round-to-nearest-even
    return (unsigned short)(u >> 16);
}

// A-fragment (16x32 bf16): lane m=l%16 is the row; cs=l/16 selects the K
// half: elems 0..15 <- kb+cs*16+0..15 (contiguous).
__device__ __forceinline__ v16bf ld_fragA(const unsigned short* rowp, int cs) {
    FragAB f;
    f.q[0] = *reinterpret_cast<const u32x4*>(rowp + cs * 16);
    f.q[1] = *reinterpret_cast<const u32x4*>(rowp + cs * 16 + 8);
    return f.v;
}
// B-fragment (32x16 bf16, B = W^T): lane n=l%16 is the column; lanes 0-15
// hold K=kb+0..15, lanes 16-31 hold K=kb+16..31 -> 16 contiguous halves.
__device__ __forceinline__ v16bf ld_fragB(const unsigned short* p) {
    FragAB f;
    f.q[0] = *reinterpret_cast<const u32x4*>(p);
    f.q[1] = *reinterpret_cast<const u32x4*>(p + 8);
    return f.v;
}

__device__ __forceinline__ v8f wmma_bf16(v16bf a, v16bf b, v8f c) {
    return __builtin_amdgcn_wmma_f32_16x16x32_bf16(false, a, false, b,
                                                   (short)0, c, false, false);
}

// ASYNCcnt-tracked global->LDS copy, 16B per lane (gfx1250 VFLAT async op).
// ldsOff = workgroup-relative LDS byte offset (low 32 bits of generic ptr,
// ISA 10.2: LDS aperture truncates flat address to addr[31:0]).
__device__ __forceinline__ void async_g2l_b128(unsigned ldsOff, const void* gaddr) {
    asm volatile("global_load_async_to_lds_b128 %0, %1, off"
                 :: "v"(ldsOff), "v"(gaddr) : "memory");
}
__device__ __forceinline__ void wait_async0() {
    asm volatile("s_wait_asynccnt 0x0" ::: "memory");
}

// ---------------------------------------------------------------------------
// f32 -> bf16 cast (weights)
// ---------------------------------------------------------------------------
__global__ void cast_bf16_kernel(const float* __restrict__ in,
                                 unsigned short* __restrict__ out, int n) {
    int i = blockIdx.x * 256 + threadIdx.x;
    if (i < n) out[i] = f2bf(in[i]);
}

// ---------------------------------------------------------------------------
// LayerNorm over EMBED, optional roll of the token index, bf16 out.
// out[b, j, :] = LN(x[b, (j+shift)&4095, :]) * g + be
// ---------------------------------------------------------------------------
__global__ __launch_bounds__(256) void ln_kernel(
    const float* __restrict__ x, const float* __restrict__ g,
    const float* __restrict__ be, unsigned short* __restrict__ out, int shift) {
    __shared__ float red[256];
    const int tid = threadIdx.x;
    const int j   = blockIdx.x;
    const int b   = j >> 12;
    const int pos = j & 4095;
    const int sp  = (pos + shift) & 4095;
    const float* xin = x + ((size_t)b * SEQL + sp) * EMBED;

    float v[4], s = 0.f;
    for (int i = 0; i < 4; ++i) { v[i] = xin[tid + i * 256]; s += v[i]; }
    red[tid] = s; __syncthreads();
    for (int o = 128; o > 0; o >>= 1) {
        if (tid < o) red[tid] += red[tid + o];
        __syncthreads();
    }
    const float mean = red[0] * (1.f / EMBED);
    __syncthreads();
    float d2 = 0.f;
    for (int i = 0; i < 4; ++i) { float d = v[i] - mean; d2 += d * d; }
    red[tid] = d2; __syncthreads();
    for (int o = 128; o > 0; o >>= 1) {
        if (tid < o) red[tid] += red[tid + o];
        __syncthreads();
    }
    const float inv = rsqrtf(red[0] * (1.f / EMBED) + 1e-5f);
    unsigned short* op = out + (size_t)j * EMBED;
    for (int i = 0; i < 4; ++i) {
        int c = tid + i * 256;
        op[c] = f2bf((v[i] - mean) * inv * g[c] + be[c]);
    }
}

// ---------------------------------------------------------------------------
// Generic bf16 GEMM:  out[m,n] = sum_k A[m,k] * W[n,k] + bias[n]  (+ epilogue)
// Block tile 128x128, 8 waves (4Mx2N) each computing 32x64 via 2x4 WMMA frags.
// Tiles staged global->LDS with async b128 copies, double-buffered (ASYNCcnt).
// MODE 0: bf16 store (qkv)
// MODE 1: proj: un-roll(+SHIFT) token remap, + residual x, f32 store (-> x1)
// MODE 2: exact GELU, bf16 store (mlp hidden)
// MODE 3: + residual x1 (same buffer), f32 store (final)
// ---------------------------------------------------------------------------
template <int MODE>
__global__ __launch_bounds__(256) void gemm_bf16_kernel(
    const unsigned short* __restrict__ A,   // M x K  (bf16)
    const unsigned short* __restrict__ Wm,  // N x K  (bf16)
    const float* __restrict__ bias,         // N
    const float* __restrict__ resid,        // residual source (modes 1,3)
    float* __restrict__ outf,               // f32 out (modes 1,3)
    unsigned short* __restrict__ outb,      // bf16 out (modes 0,2)
    int M, int N, int K) {
    constexpr int LDT   = 48;               // LDS tile row stride (halves)
    constexpr int ATILE = 128 * LDT;        // halves
    constexpr int BUFSZ = 2 * ATILE;        // A tile + B tile per buffer
    __shared__ unsigned short tiles[2 * BUFSZ];

    const int tid  = threadIdx.x;
    const int lane = tid & 31;
    const int wave = tid >> 5;
    const int wm   = wave >> 1;             // wave grid 4(M) x 2(N)
    const int wn   = wave & 1;
    const int l16  = lane & 15;
    const int cs   = lane >> 4;
    const int mBase = blockIdx.x * 128;
    const int nBase = blockIdx.y * 128;

    FragC acc[2][4];
    for (int i = 0; i < 2; ++i)
        for (int j = 0; j < 4; ++j)
            for (int r = 0; r < 8; ++r) acc[i][j].f[r] = 0.f;

    // async stage of one 128x32 A tile + one 128x32 B tile into buffer `buf`
    auto stage = [&](int buf, int kb) {
        const unsigned base = (unsigned)(uintptr_t)&tiles[buf * BUFSZ];
        for (int it = 0; it < 2; ++it) {
            const int idx = tid + it * 256;         // 0..511
            const int r   = idx >> 2;
            const int c8  = (idx & 3) * 8;
            async_g2l_b128(base + (unsigned)(r * LDT + c8) * 2,
                           A + (size_t)(mBase + r) * K + kb + c8);
            async_g2l_b128(base + (unsigned)(ATILE + r * LDT + c8) * 2,
                           Wm + (size_t)(nBase + r) * K + kb + c8);
        }
    };

    stage(0, 0);
    wait_async0();
    __syncthreads();

    int cur = 0;
    for (int kb = 0; kb < K; kb += 32) {
        if (kb + 32 < K) stage(cur ^ 1, kb + 32);   // prefetch next tile (async)
        if (kb + 64 < K)                            // -> global_prefetch_b8
            __builtin_prefetch(A + (size_t)(mBase + (tid >> 2)) * K + kb + 64, 0, 1);

        const unsigned short* As = &tiles[cur * BUFSZ];
        const unsigned short* Bs = &tiles[cur * BUFSZ + ATILE];
        v16bf af[2], bf[4];
        for (int i = 0; i < 2; ++i)
            af[i] = ld_fragA(&As[(wm * 32 + i * 16 + l16) * LDT], cs);
        for (int j = 0; j < 4; ++j)
            bf[j] = ld_fragB(&Bs[(wn * 64 + j * 16 + l16) * LDT + cs * 16]);
        for (int i = 0; i < 2; ++i)
            for (int j = 0; j < 4; ++j)
                acc[i][j].v = wmma_bf16(af[i], bf[j], acc[i][j].v);

        wait_async0();          // next tile landed in LDS
        __syncthreads();        // and everyone is done reading `cur`
        cur ^= 1;
    }

    for (int i = 0; i < 2; ++i) {
        for (int j = 0; j < 4; ++j) {
            const int nCol = nBase + wn * 64 + j * 16 + l16;
            const float bv = bias[nCol];
            for (int r = 0; r < 8; ++r) {
                const int mRow = mBase + wm * 32 + i * 16 + cs * 8 + r;
                float v = acc[i][j].f[r] + bv;
                if (MODE == 0) {
                    outb[(size_t)mRow * N + nCol] = f2bf(v);
                } else if (MODE == 1) {
                    const int b = mRow >> 12, jr = mRow & 4095;
                    const int l = (jr + SHIFTA) & 4095;          // roll(+SHIFT)
                    const size_t o = ((size_t)b * SEQL + l) * EMBED + nCol;
                    outf[o] = resid[o] + v;
                } else if (MODE == 2) {
                    float g = 0.5f * v * (1.f + erff(v * 0.70710678118f));
                    outb[(size_t)mRow * N + nCol] = f2bf(g);
                } else { // MODE == 3
                    const size_t o = (size_t)mRow * N + nCol;
                    outf[o] = resid[o] + v;
                }
            }
        }
    }
}

// ---------------------------------------------------------------------------
// Shifted-window attention, one block per (window, head).
// qkv layout: token-major [32768, 3072] bf16 (q | k | v per head).
// Phase B: S = q*k^T * scale + rpb + shift-mask   (WMMA, -> LDS f32)
// Phase C: row softmax (wave32 shfl reductions) -> P bf16 (aliases S)
// Phase D: ctx = P * v  (WMMA; v transposed through LDS)
// ---------------------------------------------------------------------------
__device__ __forceinline__ int regionOf(int pos) {          // mask region ids
    return pos < (SEQL - WIN) ? 0 : (pos < (SEQL - SHIFTA) ? 1 : 2);
}

#define S_STRIDE  132   // f32 elems per S row
#define P_STRIDE  264   // halves per P row (aliases S memory)
#define VT_STRIDE 136   // halves per vT row
#define ATTN_SMEM (WIN * S_STRIDE * 4 + HD * VT_STRIDE * 2)

__global__ __launch_bounds__(256) void attn_kernel(
    const unsigned short* __restrict__ qkv,   // [NTOK, 3*EMBED] bf16
    const float* __restrict__ rpb,            // [2*WIN-1, HEADS]
    unsigned short* __restrict__ ctx) {       // [NTOK, EMBED] bf16
    extern __shared__ char smem[];
    float*          S  = reinterpret_cast<float*>(smem);
    unsigned short* P  = reinterpret_cast<unsigned short*>(smem);
    unsigned short* vT = reinterpret_cast<unsigned short*>(smem + WIN * S_STRIDE * 4);

    const int w    = blockIdx.x >> 4;         // global window 0..255
    const int h    = blockIdx.x & 15;
    const int iw   = w & (NWIN - 1);          // window within batch
    const int tid  = threadIdx.x;
    const int lane = tid & 31;
    const int wave = tid >> 5;
    const int l16  = lane & 15;
    const int cs   = lane >> 4;

    const unsigned short* qbase = qkv + (size_t)w * WIN * (3 * EMBED) + h * HD;
    const unsigned short* kbase = qbase + EMBED;
    const unsigned short* vbase = qbase + 2 * EMBED;

    // stage v transposed: vT[d][p] = v[p][d]
    for (int idx = tid; idx < WIN * (HD / 8); idx += 256) {
        const int p = idx >> 3, c = idx & 7;
        union { u32x4 q; unsigned short s[8]; } t;
        t.q = *reinterpret_cast<const u32x4*>(vbase + (size_t)p * (3 * EMBED) + c * 8);
        for (int e = 0; e < 8; ++e) vT[(c * 8 + e) * VT_STRIDE + p] = t.s[e];
    }

    // ---- Phase B: S strip (16 rows per wave) --------------------------------
    const int m0 = wave * 16;
    FragC sacc[8];
    for (int j = 0; j < 8; ++j)
        for (int r = 0; r < 8; ++r) sacc[j].f[r] = 0.f;

    for (int kb = 0; kb < HD; kb += 32) {
        const v16bf a = ld_fragA(qbase + (size_t)(m0 + l16) * (3 * EMBED) + kb, cs);
        for (int j = 0; j < 8; ++j) {
            const v16bf b =
                ld_fragB(kbase + (size_t)(j * 16 + l16) * (3 * EMBED) + kb + cs * 16);
            sacc[j].v = wmma_bf16(a, b, sacc[j].v);
        }
    }
    for (int j = 0; j < 8; ++j) {
        const int n  = j * 16 + l16;
        const int rk = regionOf(iw * WIN + n);
        for (int r = 0; r < 8; ++r) {
            const int m  = m0 + cs * 8 + r;
            const int rq = regionOf(iw * WIN + m);
            float val = sacc[j].f[r] * SCALEA
                      + rpb[(n - m + WIN - 1) * HEADS + h]
                      + ((rq != rk) ? -100.f : 0.f);
            S[m * S_STRIDE + n] = val;
        }
    }
    __syncthreads();

    // ---- Phase C: softmax, 16 rows per wave ---------------------------------
    for (int rr = 0; rr < 16; ++rr) {
        const int row = m0 + rr;
        float v[4], mx = -3.4e38f;
        for (int i = 0; i < 4; ++i) {
            v[i] = S[row * S_STRIDE + lane + i * 32];
            mx = fmaxf(mx, v[i]);
        }
        for (int o = 16; o > 0; o >>= 1) mx = fmaxf(mx, __shfl_xor(mx, o, 32));
        float s = 0.f;
        for (int i = 0; i < 4; ++i) { v[i] = expf(v[i] - mx); s += v[i]; }
        for (int o = 16; o > 0; o >>= 1) s += __shfl_xor(s, o, 32);
        const float inv = 1.f / s;
        for (int i = 0; i < 4; ++i)
            P[row * P_STRIDE + lane + i * 32] = f2bf(v[i] * inv);
    }
    __syncthreads();

    // ---- Phase D: ctx = P * v ----------------------------------------------
    FragC oacc[4];
    for (int j = 0; j < 4; ++j)
        for (int r = 0; r < 8; ++r) oacc[j].f[r] = 0.f;
    for (int kb = 0; kb < WIN; kb += 32) {
        const v16bf a = ld_fragA(P + (m0 + l16) * P_STRIDE + kb, cs);
        for (int j = 0; j < 4; ++j) {
            const v16bf b = ld_fragB(vT + (j * 16 + l16) * VT_STRIDE + kb + cs * 16);
            oacc[j].v = wmma_bf16(a, b, oacc[j].v);
        }
    }
    for (int j = 0; j < 4; ++j) {
        const int d = j * 16 + l16;
        for (int r = 0; r < 8; ++r) {
            const int m = m0 + cs * 8 + r;
            ctx[((size_t)w * WIN + m) * EMBED + h * HD + d] = f2bf(oacc[j].f[r]);
        }
    }
}

// ---------------------------------------------------------------------------
// Host-side orchestration
// ---------------------------------------------------------------------------
extern "C" void kernel_launch(void* const* d_in, const int* in_sizes, int n_in,
                              void* d_out, int out_size, void* d_ws, size_t ws_size,
                              hipStream_t stream) {
    const float* x      = (const float*)d_in[0];
    const float* qkv_w  = (const float*)d_in[1];
    const float* qkv_b  = (const float*)d_in[2];
    const float* proj_w = (const float*)d_in[3];
    const float* proj_b = (const float*)d_in[4];
    const float* rpb    = (const float*)d_in[5];
    const float* n1w    = (const float*)d_in[6];
    const float* n1b    = (const float*)d_in[7];
    const float* n2w    = (const float*)d_in[8];
    const float* n2b    = (const float*)d_in[9];
    const float* mlp_w1 = (const float*)d_in[10];
    const float* mlp_b1 = (const float*)d_in[11];
    const float* mlp_w2 = (const float*)d_in[12];
    const float* mlp_b2 = (const float*)d_in[13];
    float* out = (float*)d_out;               // also used as x1 scratch

    // carve workspace (bf16 buffers), 256-byte aligned
    char* ws = (char*)d_ws;
    size_t off = 0;
    auto carve = [&](size_t bytes) -> unsigned short* {
        unsigned short* p = (unsigned short*)(ws + off);
        off += (bytes + 255) & ~(size_t)255;
        return p;
    };
    unsigned short* qkvw_b = carve((size_t)3 * EMBED * EMBED * 2);
    unsigned short* projw_b= carve((size_t)EMBED * EMBED * 2);
    unsigned short* w1_b   = carve((size_t)MLP_H * EMBED * 2);
    unsigned short* w2_b   = carve((size_t)EMBED * MLP_H * 2);
    unsigned short* h_b    = carve((size_t)NTOK * EMBED * 2);
    unsigned short* qkv_o  = carve((size_t)NTOK * 3 * EMBED * 2);
    unsigned short* ctx_b  = carve((size_t)NTOK * EMBED * 2);
    unsigned short* h2_b   = carve((size_t)NTOK * EMBED * 2);
    unsigned short* hid_b  = carve((size_t)NTOK * MLP_H * 2);

    // 1) weight casts
    auto cast = [&](const float* src, unsigned short* dst, int n) {
        cast_bf16_kernel<<<n / 256, 256, 0, stream>>>(src, dst, n);
    };
    cast(qkv_w,  qkvw_b, 3 * EMBED * EMBED);
    cast(proj_w, projw_b, EMBED * EMBED);
    cast(mlp_w1, w1_b, MLP_H * EMBED);
    cast(mlp_w2, w2_b, EMBED * MLP_H);

    // 2) LN1 fused with roll(-SHIFT)
    ln_kernel<<<NTOK, 256, 0, stream>>>(x, n1w, n1b, h_b, SHIFTA);

    // 3) qkv projection: [32768,1024] x [3072,1024]^T
    gemm_bf16_kernel<0><<<dim3(NTOK / 128, (3 * EMBED) / 128), 256, 0, stream>>>(
        h_b, qkvw_b, qkv_b, nullptr, nullptr, qkv_o, NTOK, 3 * EMBED, EMBED);

    // 4) windowed attention (256 windows x 16 heads)
    attn_kernel<<<BATCH * NWIN * HEADS, 256, ATTN_SMEM, stream>>>(qkv_o, rpb, ctx_b);

    // 5) output projection + roll(+SHIFT) + residual -> x1 (in d_out)
    gemm_bf16_kernel<1><<<dim3(NTOK / 128, EMBED / 128), 256, 0, stream>>>(
        ctx_b, projw_b, proj_b, x, out, nullptr, NTOK, EMBED, EMBED);

    // 6) LN2
    ln_kernel<<<NTOK, 256, 0, stream>>>(out, n2w, n2b, h2_b, 0);

    // 7) MLP up-projection + exact GELU
    gemm_bf16_kernel<2><<<dim3(NTOK / 128, MLP_H / 128), 256, 0, stream>>>(
        h2_b, w1_b, mlp_b1, nullptr, nullptr, hid_b, NTOK, MLP_H, EMBED);

    // 8) MLP down-projection + residual(x1) -> final
    gemm_bf16_kernel<3><<<dim3(NTOK / 128, EMBED / 128), 256, 0, stream>>>(
        hid_b, w2_b, mlp_b2, out, out, nullptr, NTOK, EMBED, MLP_H);

    (void)in_sizes; (void)n_in; (void)out_size; (void)ws_size;
}
